// SVDResidualLinear_9663676416607
// MI455X (gfx1250) — matrix operand order repaired and checked by reading, not compiled, not run on hardware
//
#include <hip/hip_runtime.h>

#define DIN  4096
#define DOUT 4096
#define RNK  64
#define MTOT 8192  // B*S = 8*1024

typedef __attribute__((ext_vector_type(16))) __bf16 v16bf;
typedef __attribute__((ext_vector_type(8)))  float  v8f;

static __device__ inline v16bf pack16(uint4 a, uint4 b) {
  union { uint4 u[2]; v16bf v; } t;
  t.u[0] = a; t.u[1] = b;
  return t.v;
}

static __device__ inline v8f wmma_bf16(v16bf a, v16bf b, v8f c) {
  // 8 args: (neg_a, A, neg_b, B, c_mod, C, reuse_a, reuse_b)
  return __builtin_amdgcn_wmma_f32_16x16x32_bf16(false, a, false, b, (short)0, c,
                                                 false, false);
}

// Wave-relative LDS byte address = low 32 bits of generic pointer (aperture rule).
static __device__ inline unsigned lds_lo32(const void* p) {
  return (unsigned)(uintptr_t)p;
}

// CDNA5 async copy: 16 bytes per lane, global -> LDS, tracked by ASYNCcnt.
// INST_OFFSET applies to both the LDS and the global address (ISA 15.18.3).
#define ASYNC_CP16(ldsaddr, gptr, OFF)                                        \
  asm volatile("global_load_async_to_lds_b128 %0, %1, off offset:" #OFF      \
               :: "v"(ldsaddr), "v"(gptr) : "memory")

// Copy one 64-byte row (32 bf16) global -> LDS asynchronously.
static __device__ inline void stage_row(unsigned lds, const __bf16* g) {
  ASYNC_CP16(lds, g, 0);
  ASYNC_CP16(lds, g, 16);
  ASYNC_CP16(lds, g, 32);
  ASYNC_CP16(lds, g, 48);
}

static __device__ inline void wait_async0() {
#if __has_builtin(__builtin_amdgcn_s_wait_asynccnt)
  __builtin_amdgcn_s_wait_asynccnt(0);
#else
  asm volatile("s_wait_asynccnt 0x0" ::: "memory");
#endif
}

// ---------------------------------------------------------------------------
// Kernel 1: split x (fp32) into bf16 hi/lo pair.  Each thread handles 4 elems.
// ---------------------------------------------------------------------------
__global__ __launch_bounds__(256) void split_x_kernel(
    const float* __restrict__ X, __bf16* __restrict__ Xhi, __bf16* __restrict__ Xlo) {
  size_t i = (size_t)blockIdx.x * blockDim.x + threadIdx.x;
  float4 v = ((const float4*)X)[i];
  float f[4] = {v.x, v.y, v.z, v.w};
  union { __bf16 h[4]; uint2 u; } hi, lo;
#pragma unroll
  for (int j = 0; j < 4; ++j) {
    __bf16 h = (__bf16)f[j];
    hi.h[j] = h;
    lo.h[j] = (__bf16)(f[j] - (float)h);
  }
  ((uint2*)Xhi)[i] = hi.u;
  ((uint2*)Xlo)[i] = lo.u;
}

// ---------------------------------------------------------------------------
// Kernel 2: W_eff = W + U * diag(S) * V, split into bf16 hi/lo.
// Each block computes a 64x64 tile; U tile and S*V tile staged in LDS.
// ---------------------------------------------------------------------------
__global__ __launch_bounds__(256) void prep_weight_kernel(
    const float* __restrict__ W, const float* __restrict__ U,
    const float* __restrict__ S, const float* __restrict__ V,
    __bf16* __restrict__ Whi, __bf16* __restrict__ Wlo) {
  __shared__ __align__(16) float Ut[64][64];   // [o_local][r]
  __shared__ __align__(16) float SVt[64][64];  // [r][i_local]
  const int o0 = blockIdx.y * 64;
  const int i0 = blockIdx.x * 64;
  const int t  = threadIdx.x;
  const int lr = t >> 2;          // 0..63
  const int lc = (t & 3) * 16;    // 0,16,32,48

  {  // load U tile (rows o0..o0+63, all 64 ranks)
    const float4* src = (const float4*)(U + (size_t)(o0 + lr) * RNK + lc);
    float4* dst = (float4*)&Ut[lr][lc];
#pragma unroll
    for (int j = 0; j < 4; ++j) dst[j] = src[j];
  }
  {  // load V tile scaled by S
    const float s = S[lr];
    const float4* src = (const float4*)(V + (size_t)lr * DIN + i0 + lc);
#pragma unroll
    for (int j = 0; j < 4; ++j) {
      float4 v4 = src[j];
      v4.x *= s; v4.y *= s; v4.z *= s; v4.w *= s;
      *(float4*)&SVt[lr][lc + j * 4] = v4;
    }
  }
  __syncthreads();

  const int ol = t >> 2;         // output row in tile
  const int ib = (t & 3) * 16;   // 16 output cols
  float acc[16];
  {
    const float4* w4 = (const float4*)(W + (size_t)(o0 + ol) * DIN + i0 + ib);
#pragma unroll
    for (int j = 0; j < 4; ++j) {
      float4 wv = w4[j];
      acc[j * 4 + 0] = wv.x; acc[j * 4 + 1] = wv.y;
      acc[j * 4 + 2] = wv.z; acc[j * 4 + 3] = wv.w;
    }
  }
#pragma unroll 4
  for (int r = 0; r < RNK; ++r) {
    const float u = Ut[ol][r];
#pragma unroll
    for (int c = 0; c < 16; ++c) acc[c] += u * SVt[r][ib + c];
  }
  __bf16* hrow = Whi + (size_t)(o0 + ol) * DIN + i0 + ib;
  __bf16* lrow = Wlo + (size_t)(o0 + ol) * DIN + i0 + ib;
#pragma unroll
  for (int c = 0; c < 16; ++c) {
    const float w = acc[c];
    const __bf16 h = (__bf16)w;
    hrow[c] = h;
    lrow[c] = (__bf16)(w - (float)h);
  }
}

// ---------------------------------------------------------------------------
// Fragment loaders from LDS (plane layout: [rows][32 k] bf16).
// A layout (ISA 7.12.2): lane&15 = M row; lane>=16 takes K {8..15,24..31}.
// B layout: lane&15 = N col; lane>=16 takes K {8..15,24..31} (same as A).
// ---------------------------------------------------------------------------
static __device__ inline v16bf fragA(const __bf16* plane, int r0, int lane) {
  const int row = r0 + (lane & 15);
  const int h = lane >> 4;
  const uint4 x0 = *(const uint4*)(plane + row * 32 + h * 8);
  const uint4 x1 = *(const uint4*)(plane + row * 32 + h * 8 + 16);
  return pack16(x0, x1);
}
static __device__ inline v16bf fragB(const __bf16* plane, int c0, int lane) {
  const int row = c0 + (lane & 15);
  const __bf16* q = plane + row * 32 + (lane >> 4) * 8;
  return pack16(((const uint4*)q)[0], ((const uint4*)q)[2]);
}

// ---------------------------------------------------------------------------
// Kernel 3: GEMM  Y[m,n] = sum_k Xeff[m,k]*Weff[n,k] + bias[n]
// bf16x3: acc += Ahi*Bhi + Ahi*Blo + Alo*Bhi  (fp32 WMMA accumulators)
// Block 128x128, 128 threads = 4 waves, each wave a 64x64 tile (4x4 WMMA tiles).
// K-step 32, double-buffered LDS filled by GLOBAL_LOAD_ASYNC_TO_LDS_B128.
// LDS layout per buffer (bytes): Ah 0 | Al 8192 | Bh 16384 | Bl 24576 (32 KB).
// ---------------------------------------------------------------------------
__global__ __launch_bounds__(128) void gemm_kernel(
    const __bf16* __restrict__ Xhi, const __bf16* __restrict__ Xlo,
    const __bf16* __restrict__ Whi, const __bf16* __restrict__ Wlo,
    const float* __restrict__ bias, float* __restrict__ Y) {
  __shared__ __align__(16) __bf16 smem[2][16384];  // 2 x 32 KB

  const int t    = threadIdx.x;      // 0..127
  const int lane = t & 31;
  const int wave = t >> 5;           // 0..3
  const int m0 = blockIdx.y * 128;
  const int n0 = blockIdx.x * 128;
  const int wm = (wave & 1) * 64;    // wave's row offset in block
  const int wn = (wave >> 1) * 64;   // wave's col offset in block

  // Async staging: thread t owns full 32-bf16 row t of each plane.
  const __bf16* gAh = Xhi + (size_t)(m0 + t) * DIN;
  const __bf16* gAl = Xlo + (size_t)(m0 + t) * DIN;
  const __bf16* gBh = Whi + (size_t)(n0 + t) * DIN;
  const __bf16* gBl = Wlo + (size_t)(n0 + t) * DIN;
  const unsigned ldsBase = lds_lo32(&smem[0][0]);
  const unsigned rowOff  = (unsigned)t * 64u;  // 32 bf16 = 64 bytes per row

  v8f acc[4][4];
  const v8f vzero = {0.f, 0.f, 0.f, 0.f, 0.f, 0.f, 0.f, 0.f};
#pragma unroll
  for (int i = 0; i < 4; ++i)
#pragma unroll
    for (int j = 0; j < 4; ++j) acc[i][j] = vzero;

  // Prologue: async-fill buffer 0 with k-block 0.
  {
    const unsigned base = ldsBase;
    stage_row(base +         rowOff, gAh);
    stage_row(base +  8192 + rowOff, gAl);
    stage_row(base + 16384 + rowOff, gBh);
    stage_row(base + 24576 + rowOff, gBl);
  }
  wait_async0();
  __syncthreads();

  const int NK = DIN / 32;  // 128
  for (int kt = 0; kt < NK; ++kt) {
    const int buf = kt & 1;
    if (kt + 1 < NK) {  // async-fill the other buffer with the next k-block
      const int k = (kt + 1) * 32;
      const unsigned base = ldsBase + (unsigned)(buf ^ 1) * 32768u;
      stage_row(base +         rowOff, gAh + k);
      stage_row(base +  8192 + rowOff, gAl + k);
      stage_row(base + 16384 + rowOff, gBh + k);
      stage_row(base + 24576 + rowOff, gBl + k);
    }

    const __bf16* sbuf = &smem[buf][0];
    const __bf16* pAh = sbuf;
    const __bf16* pAl = sbuf + 4096;
    const __bf16* pBh = sbuf + 8192;
    const __bf16* pBl = sbuf + 12288;

    v16bf ah[4], al[4];
#pragma unroll
    for (int r = 0; r < 4; ++r) {
      ah[r] = fragA(pAh, wm + r * 16, lane);
      al[r] = fragA(pAl, wm + r * 16, lane);
    }
#pragma unroll
    for (int u = 0; u < 4; ++u) {
      const v16bf bh = fragB(pBh, wn + u * 16, lane);
      const v16bf bl = fragB(pBl, wn + u * 16, lane);
#pragma unroll
      for (int r = 0; r < 4; ++r) {
        acc[r][u] = wmma_bf16(ah[r], bh, acc[r][u]);
        acc[r][u] = wmma_bf16(ah[r], bl, acc[r][u]);
        acc[r][u] = wmma_bf16(al[r], bh, acc[r][u]);
      }
    }

    wait_async0();     // our next-buffer fills are done
    __syncthreads();   // everyone's fills visible; reads of `buf` finished
  }

  // Epilogue: C/D layout -> lane&15 = N col, VGPR v -> row v + 8*(lane>>4)
  const int col0 = n0 + wn + (lane & 15);
  const int row0 = m0 + wm + 8 * (lane >> 4);
#pragma unroll
  for (int u = 0; u < 4; ++u) {
    const int col = col0 + u * 16;
    const float bv = bias[col];
#pragma unroll
    for (int r = 0; r < 4; ++r) {
      float* yp = Y + (size_t)(row0 + r * 16) * DOUT + col;
#pragma unroll
      for (int v = 0; v < 8; ++v) {
        yp[(size_t)v * DOUT] = acc[r][u][v] + bv;
      }
    }
  }
}

// ---------------------------------------------------------------------------
extern "C" void kernel_launch(void* const* d_in, const int* in_sizes, int n_in,
                              void* d_out, int out_size, void* d_ws, size_t ws_size,
                              hipStream_t stream) {
  (void)in_sizes; (void)n_in; (void)out_size; (void)ws_size;
  const float* x    = (const float*)d_in[0];
  const float* W    = (const float*)d_in[1];
  const float* U    = (const float*)d_in[2];
  const float* S    = (const float*)d_in[3];
  const float* V    = (const float*)d_in[4];
  const float* bias = (const float*)d_in[5];
  float* Y = (float*)d_out;

  // Workspace layout (bf16): Whi | Wlo | Xhi | Xlo  (~201 MB total)
  char* ws = (char*)d_ws;
  const size_t wbytes = (size_t)DOUT * DIN * sizeof(__bf16);
  const size_t xbytes = (size_t)MTOT * DIN * sizeof(__bf16);
  __bf16* Whi = (__bf16*)(ws);
  __bf16* Wlo = (__bf16*)(ws + wbytes);
  __bf16* Xhi = (__bf16*)(ws + 2 * wbytes);
  __bf16* Xlo = (__bf16*)(ws + 2 * wbytes + xbytes);

  // 1) split x to bf16 hi/lo
  {
    const int nblk = (int)(((size_t)MTOT * DIN / 4) / 256);  // 32768
    split_x_kernel<<<nblk, 256, 0, stream>>>(x, Xhi, Xlo);
  }
  // 2) fold low-rank residual into W, split to bf16 hi/lo
  {
    dim3 g(DIN / 64, DOUT / 64);
    prep_weight_kernel<<<g, 256, 0, stream>>>(W, U, S, V, Whi, Wlo);
  }
  // 3) bf16x3 WMMA GEMM with fused bias, async-LDS staged
  {
    dim3 g(DOUT / 128, MTOT / 128);  // n fastest -> x stays hot in L2
    gemm_kernel<<<g, 128, 0, stream>>>(Xhi, Xlo, Whi, Wlo, bias, Y);
  }
}
